// MoETransformerBlock_51049981281423
// MI455X (gfx1250) — compile-verified, matrix operand order checked by reading, not run on hardware
//
#include <hip/hip_runtime.h>
#include <cstdint>
#include <cstddef>

typedef unsigned short u16;
typedef __attribute__((ext_vector_type(16))) __bf16 v16bf;
typedef __attribute__((ext_vector_type(8)))  float  v8f;

namespace {
constexpr int kS = 2048, kD = 2048, kH = 32, kG = 8, kHD = 64;
constexpr int kE = 8, kF = 768, kT = 2048;
constexpr float kEPS = 1e-6f;
}

// ---------- helpers ----------
__device__ __forceinline__ u16 f2bf(float f) {
  unsigned u = __float_as_uint(f);
  unsigned r = u + 0x7FFFu + ((u >> 16) & 1u);   // round-to-nearest-even
  return (u16)(r >> 16);
}
__device__ __forceinline__ float bf2f(u16 h) { return __uint_as_float(((unsigned)h) << 16); }

union Frag { v16bf v; u16 u[16]; uint4 q[2]; };

__device__ __forceinline__ v8f wmma_bf16(const Frag& a, const Frag& b, v8f c) {
  return __builtin_amdgcn_wmma_f32_16x16x32_bf16(false, a.v, false, b.v, (short)0, c,
                                                 false, false);
}
__device__ __forceinline__ float wave_red_sum(float v) {
  for (int off = 16; off; off >>= 1) v += __shfl_xor(v, off, 32);
  return v;
}
__device__ __forceinline__ float half_red_max(float v) {   // within 16-lane half
  for (int off = 8; off; off >>= 1) v = fmaxf(v, __shfl_xor(v, off, 32));
  return v;
}
__device__ __forceinline__ float half_red_sum(float v) {
  for (int off = 8; off; off >>= 1) v += __shfl_xor(v, off, 32);
  return v;
}
// CDNA5 async global->LDS copy (16B per lane), tracked by ASYNCcnt
__device__ __forceinline__ void async_b128_to_lds(unsigned lds_off, const void* gptr) {
  asm volatile("global_load_async_to_lds_b128 %0, %1, off"
               :: "v"(lds_off), "v"((unsigned long long)(uintptr_t)gptr) : "memory");
}

// ---------- elementwise kernels ----------
// vectorized fp32 -> bf16 (4 elements per thread); n must be a multiple of 4
__global__ void f2bf_kernel(const float* __restrict__ in, u16* __restrict__ out, int n4) {
  int i = blockIdx.x * 256 + threadIdx.x;
  if (i < n4) {
    float4 f = ((const float4*)in)[i];
    union { u16 u[4]; uint2 v; } o;
    o.u[0] = f2bf(f.x); o.u[1] = f2bf(f.y); o.u[2] = f2bf(f.z); o.u[3] = f2bf(f.w);
    ((uint2*)out)[i] = o.v;
  }
}

__global__ __launch_bounds__(256) void rmsnorm_kernel(const float* __restrict__ x,
                                                      const float* __restrict__ sc,
                                                      u16* __restrict__ out) {
  int t = blockIdx.x, tid = threadIdx.x, lane = tid & 31, wid = tid >> 5;
  float ss = 0.f;
  for (int d = tid; d < kD; d += 256) { float v = x[(size_t)t * kD + d]; ss += v * v; }
  ss = wave_red_sum(ss);
  __shared__ float red[8];
  if (lane == 0) red[wid] = ss;
  __syncthreads();
  float tot = 0.f;
  for (int i = 0; i < 8; ++i) tot += red[i];
  float inv = rsqrtf(tot / (float)kD + kEPS);
  for (int d = tid; d < kD; d += 256)
    out[(size_t)t * kD + d] = f2bf(x[(size_t)t * kD + d] * inv * sc[d]);
}

// per-head RMSNorm + RoPE; in: [S, nh*64] f32; out: [nh, S, 64] bf16
__global__ __launch_bounds__(256) void qk_rope_kernel(const float* __restrict__ in,
                                                      const float* __restrict__ sc,
                                                      const float* __restrict__ cosT,
                                                      const float* __restrict__ sinT,
                                                      u16* __restrict__ out, int nh) {
  int gw = blockIdx.x * 8 + (threadIdx.x >> 5);
  int lane = threadIdx.x & 31;
  int s = gw / nh, h = gw % nh;
  const float* row = in + ((size_t)s * nh + h) * kHD;
  float lo = row[lane], hi = row[lane + 32];
  float ss = wave_red_sum(lo * lo + hi * hi);
  float inv = rsqrtf(ss / (float)kHD + kEPS);
  float nlo = lo * inv * sc[lane], nhi = hi * inv * sc[lane + 32];
  float c0 = cosT[(size_t)s * kHD + lane],      s0 = sinT[(size_t)s * kHD + lane];
  float c1 = cosT[(size_t)s * kHD + lane + 32], s1 = sinT[(size_t)s * kHD + lane + 32];
  float olo = nlo * c0 - nhi * s0;
  float ohi = nhi * c1 + nlo * s1;
  out[((size_t)h * kS + s) * kHD + lane]      = f2bf(olo);
  out[((size_t)h * kS + s) * kHD + lane + 32] = f2bf(ohi);
}

// v f32 [S, G*64] -> vt bf16 [G, 64, S]
__global__ void v_transpose_kernel(const float* __restrict__ vin, u16* __restrict__ vt) {
  int i = blockIdx.x * 256 + threadIdx.x;
  int s = i >> 9, rem = i & 511, g = rem >> 6, hd = rem & 63;
  vt[((size_t)g * kHD + hd) * kS + s] = f2bf(vin[i]);
}

// ---------- generic bf16 WMMA GEMM (plain / gathered / scattered) ----------
// Double-buffered LDS; A tile staged with async global->LDS (ASYNCcnt),
// B tile staged through registers (transpose scatter), one tile ahead.
#define BM 64
#define BN 64
#define BK 32
__global__ __launch_bounds__(128) void gemm_bf16_kernel(
    const u16* __restrict__ A, const u16* __restrict__ Bb, float* __restrict__ Cb,
    const float* __restrict__ residual,
    const int* __restrict__ a_map, int a_map_shift,
    const int* __restrict__ c_map, const float* __restrict__ c_scale,
    const int* __restrict__ cnt,
    int M, int N, int K,
    long long strideB_e, int map_stride_e, int arow_off_e, int crow_off_e) {
  int e = blockIdx.z;
  int rows = cnt ? cnt[e] : M;
  int m0 = blockIdx.y * BM;
  if (m0 >= rows) return;
  int n0 = blockIdx.x * BN;
  const u16* B = Bb + (size_t)((long long)e * strideB_e);
  const int* amap = a_map ? a_map + (size_t)e * map_stride_e : nullptr;
  const int* cmap = c_map ? c_map + (size_t)e * map_stride_e : nullptr;
  const float* csc = c_scale ? c_scale + (size_t)e * map_stride_e : nullptr;

  __shared__ __align__(16) u16 As[2][BM][BK + 8];   // 80B rows (16B aligned)
  __shared__ __align__(16) u16 Bs[2][BN][BK + 8];   // transposed: [n][k]
  __shared__ int rowmap[BM];

  int tid = threadIdx.x, lane = tid & 31, wid = tid >> 5;

  // A source row per tile row: computed once per block
  for (int r = tid; r < BM; r += 128) {
    int gr = m0 + r;
    int arow = e * arow_off_e;
    if (gr < rows) arow += (amap ? (amap[gr] >> a_map_shift) : gr);
    rowmap[r] = arow;
  }
  __syncthreads();

  // per-thread staging coordinates (static w.r.t. K)
  int ar0 = tid >> 2, akv0 = (tid & 3) * 8;                  // A vec 0
  int ar1 = (tid + 128) >> 2, akv1 = ((tid + 128) & 3) * 8;  // A vec 1
  int bk0 = tid >> 3, bn0 = (tid & 7) * 8;                   // B vec 0
  int bk1 = (tid + 128) >> 3, bn1 = ((tid + 128) & 7) * 8;   // B vec 1

  // issue async A-tile (2 x b128 per thread => 2 ASYNCcnt per wave... per lane pair)
  auto stage_a = [&](int k0, int b) {
    async_b128_to_lds((unsigned)(uintptr_t)&As[b][ar0][akv0],
                      &A[(size_t)rowmap[ar0] * K + k0 + akv0]);
    async_b128_to_lds((unsigned)(uintptr_t)&As[b][ar1][akv1],
                      &A[(size_t)rowmap[ar1] * K + k0 + akv1]);
  };
  uint4 rb0, rb1;
  auto load_b = [&](int k0) {
    rb0 = *(const uint4*)&B[(size_t)(k0 + bk0) * N + n0 + bn0];
    rb1 = *(const uint4*)&B[(size_t)(k0 + bk1) * N + n0 + bn1];
  };
  auto store_b = [&](int b) {
    union { uint4 q; u16 u[8]; } d0, d1;
    d0.q = rb0; d1.q = rb1;
#pragma unroll
    for (int j = 0; j < 8; ++j) Bs[b][bn0 + j][bk0] = d0.u[j];
#pragma unroll
    for (int j = 0; j < 8; ++j) Bs[b][bn1 + j][bk1] = d1.u[j];
  };

  v8f acc[4];
#pragma unroll
  for (int t = 0; t < 4; ++t)
#pragma unroll
    for (int i = 0; i < 8; ++i) acc[t][i] = 0.f;

  int fr = lane & 15;
  int kb = (lane & 16) ? 8 : 0;
  int half = (lane >> 4) & 1;

  stage_a(0, 0);   // async tile 0 in flight
  load_b(0);
  int cur = 0;
  for (int k0 = 0; k0 < K; k0 += BK) {
    store_b(cur);
    if (k0 + BK < K) {
      stage_a(k0 + BK, cur ^ 1);   // next A tile in flight during compute
      load_b(k0 + BK);             // next B tile latency hidden under compute
      asm volatile("s_wait_asynccnt 0x2" ::: "memory");  // current A tile done
    } else {
      asm volatile("s_wait_asynccnt 0x0" ::: "memory");
    }
    __syncthreads();
    Frag af;
    af.q[0] = *(const uint4*)&As[cur][wid * 16 + fr][kb];
    af.q[1] = *(const uint4*)&As[cur][wid * 16 + fr][kb + 16];
#pragma unroll
    for (int t = 0; t < 4; ++t) {
      Frag bf_;
      bf_.q[0] = *(const uint4*)&Bs[cur][t * 16 + fr][kb];
      bf_.q[1] = *(const uint4*)&Bs[cur][t * 16 + fr][kb + 16];
      acc[t] = wmma_bf16(af, bf_, acc[t]);
    }
    __syncthreads();
    cur ^= 1;
  }
#pragma unroll
  for (int t = 0; t < 4; ++t)
#pragma unroll
    for (int i = 0; i < 8; ++i) {
      int gr = m0 + wid * 16 + i + half * 8;
      if (gr >= rows) continue;
      int col = n0 + t * 16 + fr;
      int crow = cmap ? cmap[gr] : gr + e * crow_off_e;
      float v = acc[t][i] * (csc ? csc[gr] : 1.f);
      if (residual) v += residual[(size_t)crow * N + col];
      Cb[(size_t)crow * N + col] = v;
    }
}

// ---------- flash attention: 1 wave = 16 queries x full head ----------
__global__ __launch_bounds__(32) void flash_attn_kernel(const u16* __restrict__ Qb,
                                                        const u16* __restrict__ Kb,
                                                        const u16* __restrict__ Vtb,
                                                        u16* __restrict__ ctx) {
  __shared__ __align__(16) u16 Pst[16][40];
  const int q0 = blockIdx.x * 16;
  const int h = blockIdx.y;
  const int g = h / (kH / kG);
  const int lane = threadIdx.x;
  const int r = lane & 15;
  const int kb = (lane & 16) ? 8 : 0;
  const int half = (lane >> 4) & 1;

  const u16* Qh = Qb + ((size_t)h * kS + q0) * kHD;
  Frag aq[2];
#pragma unroll
  for (int c = 0; c < 2; ++c) {
    aq[c].q[0] = *(const uint4*)&Qh[(size_t)r * kHD + c * 32 + kb];
    aq[c].q[1] = *(const uint4*)&Qh[(size_t)r * kHD + c * 32 + kb + 16];
  }
  const u16* Kg = Kb + (size_t)g * kS * kHD;
  const u16* Vg = Vtb + (size_t)g * kHD * kS;

  v8f acc[4];
  float rm[8], rl[8];
#pragma unroll
  for (int i = 0; i < 8; ++i) { rm[i] = -3e38f; rl[i] = 0.f; }
#pragma unroll
  for (int t = 0; t < 4; ++t)
#pragma unroll
    for (int i = 0; i < 8; ++i) acc[t][i] = 0.f;

  for (int j0 = 0; j0 < q0 + 16; j0 += 32) {
    v8f s0, s1;
#pragma unroll
    for (int i = 0; i < 8; ++i) { s0[i] = 0.f; s1[i] = 0.f; }
#pragma unroll
    for (int c = 0; c < 2; ++c) {
      Frag b0, b1;
      b0.q[0] = *(const uint4*)&Kg[(size_t)(j0 + r) * kHD + c * 32 + kb];
      b0.q[1] = *(const uint4*)&Kg[(size_t)(j0 + r) * kHD + c * 32 + kb + 16];
      b1.q[0] = *(const uint4*)&Kg[(size_t)(j0 + 16 + r) * kHD + c * 32 + kb];
      b1.q[1] = *(const uint4*)&Kg[(size_t)(j0 + 16 + r) * kHD + c * 32 + kb + 16];
      s0 = wmma_bf16(aq[c], b0, s0);
      s1 = wmma_bf16(aq[c], b1, s1);
    }
    const bool edge = (j0 + 31 > q0);
#pragma unroll
    for (int i = 0; i < 8; ++i) {
      float a = s0[i] * 0.125f;          // 1/sqrt(64)
      float b = s1[i] * 0.125f;
      if (edge) {
        int qg = q0 + i + half * 8;
        if (j0 + r > qg)      a = -1e30f;
        if (j0 + 16 + r > qg) b = -1e30f;
      }
      float bm = half_red_max(fmaxf(a, b));
      float nm = fmaxf(rm[i], bm);
      float al = __expf(rm[i] - nm);
      rm[i] = nm;
      float pa = __expf(a - nm);
      float pb = __expf(b - nm);
      float bs = half_red_sum(pa + pb);
      rl[i] = rl[i] * al + bs;
#pragma unroll
      for (int t = 0; t < 4; ++t) acc[t][i] *= al;
      Pst[i + half * 8][r]      = f2bf(pa);
      Pst[i + half * 8][16 + r] = f2bf(pb);
    }
    __syncthreads();
    Frag ap;
    ap.q[0] = *(const uint4*)&Pst[r][kb];
    ap.q[1] = *(const uint4*)&Pst[r][kb + 16];
#pragma unroll
    for (int t = 0; t < 4; ++t) {
      Frag bv;
      bv.q[0] = *(const uint4*)&Vg[(size_t)(t * 16 + r) * kS + j0 + kb];
      bv.q[1] = *(const uint4*)&Vg[(size_t)(t * 16 + r) * kS + j0 + kb + 16];
      acc[t] = wmma_bf16(ap, bv, acc[t]);
    }
    __syncthreads();
  }
#pragma unroll
  for (int t = 0; t < 4; ++t)
#pragma unroll
    for (int i = 0; i < 8; ++i) {
      int gr = q0 + i + half * 8;
      int col = h * kHD + t * 16 + r;
      ctx[(size_t)gr * (kH * kHD) + col] = f2bf(acc[t][i] / rl[i]);
    }
}

// ---------- MoE routing ----------
__global__ void zero_cnt_kernel(int* cnt) {
  if (threadIdx.x < kE) cnt[threadIdx.x] = 0;
}

__global__ __launch_bounds__(256) void router_kernel(const u16* __restrict__ h2,
                                                     const float* __restrict__ Wr,
                                                     int* __restrict__ cnt,
                                                     int* __restrict__ entries,
                                                     float* __restrict__ wl) {
  int t = blockIdx.x, wid = threadIdx.x >> 5, lane = threadIdx.x & 31;
  float s = 0.f;
  for (int d = lane; d < kD; d += 32)
    s += bf2f(h2[(size_t)t * kD + d]) * Wr[(size_t)d * kE + wid];
  s = wave_red_sum(s);
  __shared__ float lg[kE];
  if (lane == 0) lg[wid] = s;
  __syncthreads();
  if (threadIdx.x == 0) {
    float m = lg[0];
    for (int i = 1; i < kE; ++i) m = fmaxf(m, lg[i]);
    float p[kE];
    for (int i = 0; i < kE; ++i) p[i] = __expf(lg[i] - m);
    int i0 = 0;
    for (int i = 1; i < kE; ++i) if (p[i] > p[i0]) i0 = i;
    int i1 = (i0 == 0) ? 1 : 0;
    for (int i = 0; i < kE; ++i) if (i != i0 && p[i] > p[i1]) i1 = i;
    float denom = p[i0] + p[i1];
    int p0 = atomicAdd(&cnt[i0], 1);
    entries[i0 * kT + p0] = t * 2;     wl[i0 * kT + p0] = p[i0] / denom;
    int p1 = atomicAdd(&cnt[i1], 1);
    entries[i1 * kT + p1] = t * 2 + 1; wl[i1 * kT + p1] = p[i1] / denom;
  }
}

__global__ __launch_bounds__(256) void silu_mul_kernel(const float* __restrict__ gate,
                                                       const float* __restrict__ up,
                                                       u16* __restrict__ hid,
                                                       const int* __restrict__ cnt) {
  int e = blockIdx.y, i = blockIdx.x;
  if (i >= cnt[e]) return;
  size_t base = ((size_t)e * kT + i) * kF;
  for (int f = threadIdx.x; f < kF; f += 256) {
    float g = gate[base + f], u = up[base + f];
    hid[base + f] = f2bf((g / (1.f + __expf(-g))) * u);
  }
}

__global__ void combine_kernel(const float* __restrict__ x1, const float* __restrict__ down,
                               float* __restrict__ out) {
  int i = blockIdx.x * 256 + threadIdx.x;
  int t = i >> 11, d = i & 2047;
  out[i] = x1[i] + down[(size_t)(t * 2) * kD + d] + down[(size_t)(t * 2 + 1) * kD + d];
}

// ---------- host orchestration ----------
extern "C" void kernel_launch(void* const* d_in, const int* in_sizes, int n_in,
                              void* d_out, int out_size, void* d_ws, size_t ws_size,
                              hipStream_t stream) {
  (void)in_sizes; (void)n_in; (void)out_size; (void)ws_size;
  const float* x    = (const float*)d_in[0];
  const float* cosT = (const float*)d_in[2];
  const float* sinT = (const float*)d_in[3];
  const float* n1s  = (const float*)d_in[4];
  const float* n2s  = (const float*)d_in[5];
  const float* Wq   = (const float*)d_in[6];
  const float* Wk   = (const float*)d_in[7];
  const float* Wv   = (const float*)d_in[8];
  const float* Wo   = (const float*)d_in[9];
  const float* qns  = (const float*)d_in[10];
  const float* kns  = (const float*)d_in[11];
  const float* Wr   = (const float*)d_in[12];
  const float* Wg   = (const float*)d_in[13];
  const float* Wu   = (const float*)d_in[14];
  const float* Wd   = (const float*)d_in[15];
  float* out = (float*)d_out;

  char* base = (char*)d_ws;
  size_t off = 0;
  auto alloc = [&](size_t bytes) -> void* {
    void* p = base + off;
    off += (bytes + 255) & ~(size_t)255;
    return p;
  };
  u16*   wq_bf  = (u16*)alloc((size_t)kD * kH * kHD * 2);
  u16*   wk_bf  = (u16*)alloc((size_t)kD * kG * kHD * 2);
  u16*   wv_bf  = (u16*)alloc((size_t)kD * kG * kHD * 2);
  u16*   wo_bf  = (u16*)alloc((size_t)kH * kHD * kD * 2);
  u16*   wg_bf  = (u16*)alloc((size_t)kE * kD * kF * 2);
  u16*   wu_bf  = (u16*)alloc((size_t)kE * kD * kF * 2);
  u16*   wd_bf  = (u16*)alloc((size_t)kE * kF * kD * 2);
  u16*   h1_bf  = (u16*)alloc((size_t)kS * kD * 2);
  float* qf     = (float*)alloc((size_t)kS * kH * kHD * 4);
  float* kf     = (float*)alloc((size_t)kS * kG * kHD * 4);
  float* vf     = (float*)alloc((size_t)kS * kG * kHD * 4);
  u16*   q_bf   = (u16*)alloc((size_t)kH * kS * kHD * 2);
  u16*   k_bf   = (u16*)alloc((size_t)kG * kS * kHD * 2);
  u16*   vt_bf  = (u16*)alloc((size_t)kG * kHD * kS * 2);
  u16*   ctx_bf = (u16*)alloc((size_t)kS * kH * kHD * 2);
  float* x1     = (float*)alloc((size_t)kS * kD * 4);
  u16*   h2_bf  = (u16*)alloc((size_t)kS * kD * 2);
  int*   cnt    = (int*)alloc(kE * 4);
  int*   entr   = (int*)alloc((size_t)kE * kT * 4);
  float* wl     = (float*)alloc((size_t)kE * kT * 4);
  float* gate   = (float*)alloc((size_t)kE * kT * kF * 4);
  float* up     = (float*)alloc((size_t)kE * kT * kF * 4);
  u16*   hid    = (u16*)alloc((size_t)kE * kT * kF * 2);
  float* down   = (float*)alloc((size_t)kT * 2 * kD * 4);

  auto cvt = [&](const float* src, u16* dst, size_t n) {
    int n4 = (int)(n / 4);
    f2bf_kernel<<<dim3((unsigned)((n4 + 255) / 256)), 256, 0, stream>>>(src, dst, n4);
  };
  cvt(Wq, wq_bf, (size_t)kD * kH * kHD);
  cvt(Wk, wk_bf, (size_t)kD * kG * kHD);
  cvt(Wv, wv_bf, (size_t)kD * kG * kHD);
  cvt(Wo, wo_bf, (size_t)kH * kHD * kD);
  cvt(Wg, wg_bf, (size_t)kE * kD * kF);
  cvt(Wu, wu_bf, (size_t)kE * kD * kF);
  cvt(Wd, wd_bf, (size_t)kE * kF * kD);

  auto gemm = [&](const u16* Ap, const u16* Bp, float* Cp, const float* res,
                  const int* amap, int ashift, const int* cmap, const float* cscale,
                  const int* cp, int M, int N, int K,
                  long long sBe, int mapE, int aoffE, int coffE, int ez) {
    dim3 grid(N / 64, (M + 63) / 64, ez);
    gemm_bf16_kernel<<<grid, 128, 0, stream>>>(Ap, Bp, Cp, res, amap, ashift, cmap, cscale,
                                               cp, M, N, K, sBe, mapE, aoffE, coffE);
  };

  // ---- attention sub-block ----
  rmsnorm_kernel<<<kS, 256, 0, stream>>>(x, n1s, h1_bf);
  gemm(h1_bf, wq_bf, qf, nullptr, nullptr, 0, nullptr, nullptr, nullptr,
       kS, kH * kHD, kD, 0, 0, 0, 0, 1);
  gemm(h1_bf, wk_bf, kf, nullptr, nullptr, 0, nullptr, nullptr, nullptr,
       kS, kG * kHD, kD, 0, 0, 0, 0, 1);
  gemm(h1_bf, wv_bf, vf, nullptr, nullptr, 0, nullptr, nullptr, nullptr,
       kS, kG * kHD, kD, 0, 0, 0, 0, 1);
  qk_rope_kernel<<<kS * kH / 8, 256, 0, stream>>>(qf, qns, cosT, sinT, q_bf, kH);
  qk_rope_kernel<<<kS * kG / 8, 256, 0, stream>>>(kf, kns, cosT, sinT, k_bf, kG);
  v_transpose_kernel<<<(kS * kG * kHD) / 256, 256, 0, stream>>>(vf, vt_bf);
  flash_attn_kernel<<<dim3(kS / 16, kH), 32, 0, stream>>>(q_bf, k_bf, vt_bf, ctx_bf);
  gemm(ctx_bf, wo_bf, x1, x, nullptr, 0, nullptr, nullptr, nullptr,
       kS, kD, kH * kHD, 0, 0, 0, 0, 1);

  // ---- MoE sub-block ----
  rmsnorm_kernel<<<kS, 256, 0, stream>>>(x1, n2s, h2_bf);
  zero_cnt_kernel<<<1, 32, 0, stream>>>(cnt);
  router_kernel<<<kT, 256, 0, stream>>>(h2_bf, Wr, cnt, entr, wl);
  gemm(h2_bf, wg_bf, gate, nullptr, entr, 1, nullptr, nullptr, cnt,
       kT, kF, kD, (long long)kD * kF, kT, 0, kT, kE);
  gemm(h2_bf, wu_bf, up, nullptr, entr, 1, nullptr, nullptr, cnt,
       kT, kF, kD, (long long)kD * kF, kT, 0, kT, kE);
  silu_mul_kernel<<<dim3(kT, kE), 256, 0, stream>>>(gate, up, hid, cnt);
  gemm(hid, wd_bf, down, nullptr, nullptr, 0, entr, wl, cnt,
       kT, kD, kF, (long long)kF * kD, kT, kT, 0, kE);
  combine_kernel<<<(kS * kD) / 256, 256, 0, stream>>>(x1, down, out);
}
